// GCNConv_Model_2783138808452
// MI455X (gfx1250) — matrix operand order repaired and checked by reading, not compiled, not run on hardware
//
#include <hip/hip_runtime.h>
#include <hip/hip_bf16.h>

typedef __attribute__((ext_vector_type(16))) _Float16 v16h;
typedef __attribute__((ext_vector_type(8)))  float    v8f;

#define NN      100000
#define EE      3200000
#define FIN     128
#define EMB     64
#define GG      256
#define NC      10
#define BN_EPS  1e-5f

// ---------------------------------------------------------------------------
// generic fill
__global__ void fill_f32(float* __restrict__ p, long long n, float v) {
    long long i = (long long)blockIdx.x * blockDim.x + threadIdx.x;
    if (i < n) p[i] = v;
}

// ---------------------------------------------------------------------------
// degree count: deg[dst] += 1 per edge
__global__ void count_deg(const int* __restrict__ ei, float* __restrict__ deg, int e_total) {
    int e = blockIdx.x * blockDim.x + threadIdx.x;
    if (e >= e_total) return;
    int d = ei[e_total + e];           // edge_index[1][e]
    atomicAdd(&deg[d], 1.0f);
}

__global__ void finalize_dinv(float* __restrict__ deg, int n) {
    int i = blockIdx.x * blockDim.x + threadIdx.x;
    if (i < n) deg[i] = rsqrtf(deg[i] + 1.0f);
}

// ---------------------------------------------------------------------------
// h = x @ conv_w   via V_WMMA_F32_16X16X32_F16
// Block: 256 threads = 8 waves; each wave owns a 16-row x 64-col strip.
// W (128x64 fp32) is repacked ONCE per block into a contiguous-k B-fragment VGPR
// layout (k = 16*(lane/16) + e, as for A) in LDS, so each wave pulls a fragment as one contiguous
// 32-byte LDS vector load instead of 16 strided ds_load_u16.
__global__ __launch_bounds__(256) void conv_gemm_wmma(const float* __restrict__ x,
                                                      const float* __restrict__ w,
                                                      float* __restrict__ h, int n) {
    // bfrag[kk/32][colTile][lane][elem]; elem e of lane L = W[kk + (L>>4)*16 + e][t*16 + (L&15)]
    __shared__ alignas(32) _Float16 bfrag[4][4][32][16];      // 16 KB
    int tid = threadIdx.x;
    for (int i = tid; i < 4 * 4 * 32 * 16; i += 256) {
        int elem = i & 15;
        int lane = (i >> 4) & 31;
        int t    = (i >> 9) & 3;
        int kkI  = (i >> 11) & 3;
        int col  = t * 16 + (lane & 15);
        int k    = kkI * 32 + (lane >> 4) * 16 + elem;
        bfrag[kkI][t][lane][elem] = (_Float16)w[k * EMB + col];
    }
    __syncthreads();

    const int wave = tid >> 5;
    const int lane = tid & 31;
    const int hi   = lane >> 4;                  // 0: lanes 0-15, 1: lanes 16-31
    const int rowBase = blockIdx.x * 128 + wave * 16;
    if (rowBase >= n) return;

    const int rA = min(rowBase + (lane & 15), n - 1);   // A-row (clamped for tail)
    const int col0 = lane & 15;

    v8f acc[4] = {};                             // 4 col-tiles of 16 -> 64 cols

    #pragma unroll
    for (int kkI = 0; kkI < 4; ++kkI) {
        // A fragment 16x32 f16: lanes 0-15 hold K=kk..kk+15,
        // lanes 16-31 hold K=kk+16..kk+31  (rows M = lane%16 in both halves)
        v16h a;
        const float* xr = x + (size_t)rA * FIN + kkI * 32 + hi * 16;
        const float4 x0 = *(const float4*)(xr);
        const float4 x1 = *(const float4*)(xr + 4);
        const float4 x2 = *(const float4*)(xr + 8);
        const float4 x3 = *(const float4*)(xr + 12);
        a[0]  = (_Float16)x0.x;  a[1]  = (_Float16)x0.y;
        a[2]  = (_Float16)x0.z;  a[3]  = (_Float16)x0.w;
        a[4]  = (_Float16)x1.x;  a[5]  = (_Float16)x1.y;
        a[6]  = (_Float16)x1.z;  a[7]  = (_Float16)x1.w;
        a[8]  = (_Float16)x2.x;  a[9]  = (_Float16)x2.y;
        a[10] = (_Float16)x2.z;  a[11] = (_Float16)x2.w;
        a[12] = (_Float16)x3.x;  a[13] = (_Float16)x3.y;
        a[14] = (_Float16)x3.z;  a[15] = (_Float16)x3.w;
        #pragma unroll
        for (int t = 0; t < 4; ++t) {
            v16h b = *(const v16h*)&bfrag[kkI][t][lane][0];   // contiguous LDS fragment
            acc[t] = __builtin_amdgcn_wmma_f32_16x16x32_f16(
                false, a, false, b, (short)0, acc[t], false, false);
        }
    }

    // C/D layout: VGPR i -> row i (lanes 0-15) / row i+8 (lanes 16-31), col = lane%16
    if (rowBase + 16 <= n) {                     // fast path: no per-row guard
        float* hp = h + (size_t)(rowBase + hi * 8) * EMB + col0;
        #pragma unroll
        for (int i = 0; i < 8; ++i) {
            #pragma unroll
            for (int t = 0; t < 4; ++t)
                hp[(size_t)i * EMB + t * 16] = acc[t][i];
        }
    } else {                                     // tail block only
        #pragma unroll
        for (int i = 0; i < 8; ++i) {
            int m = rowBase + i + hi * 8;
            if (m < n) {
                #pragma unroll
                for (int t = 0; t < 4; ++t)
                    h[(size_t)m * EMB + t * 16 + col0] = acc[t][i];
            }
        }
    }
}

// ---------------------------------------------------------------------------
// agg[n][c] = h[n][c] * dinv[n]^2 + conv_b[c]   (self-loop term + bias)
__global__ void init_agg(const float* __restrict__ h, const float* __restrict__ dinv,
                         const float* __restrict__ cb, float* __restrict__ agg, int n) {
    long long t = (long long)blockIdx.x * blockDim.x + threadIdx.x;
    int r = (int)(t >> 6);
    if (r >= n) return;
    int c = (int)(t & 63);
    float di = dinv[r];
    agg[(size_t)r * EMB + c] = h[(size_t)r * EMB + c] * di * di + cb[c];
}

// ---------------------------------------------------------------------------
// edge scatter: agg[dst] += h[src] * (dinv[src]*dinv[dst]); 16 lanes/edge x float4
__global__ void edge_scatter(const int* __restrict__ ei, const float* __restrict__ h,
                             const float* __restrict__ dinv, float* __restrict__ agg,
                             int e_total) {
    long long t = (long long)blockIdx.x * blockDim.x + threadIdx.x;
    long long e = t >> 4;
    if (e >= e_total) return;
    int f = (int)(t & 15) * 4;
    __builtin_prefetch(ei + e + 8192, 0, 0);            // global_prefetch_b8 on index stream
    int s = ei[e];
    int d = ei[e_total + e];
    float nrm = dinv[s] * dinv[d];
    const float4 hv = *(const float4*)(h + (size_t)s * EMB + f);
    float* ap = agg + (size_t)d * EMB + f;
    atomicAdd(ap + 0, hv.x * nrm);
    atomicAdd(ap + 1, hv.y * nrm);
    atomicAdd(ap + 2, hv.z * nrm);
    atomicAdd(ap + 3, hv.w * nrm);
}

// ---------------------------------------------------------------------------
// BatchNorm stats over N rows, 64 channels: accum[c]=sum, accum[64+c]=sumsq
__global__ void bn_stats(const float* __restrict__ a, float* __restrict__ accum, int n) {
    int c = threadIdx.x & 63;
    int rstart  = blockIdx.x * (blockDim.x >> 6) + (threadIdx.x >> 6);
    int rstride = gridDim.x * (blockDim.x >> 6);
    float s = 0.f, s2 = 0.f;
    for (int r = rstart; r < n; r += rstride) {
        float v = a[(size_t)r * EMB + c];
        s += v; s2 += v * v;
    }
    atomicAdd(&accum[c], s);
    atomicAdd(&accum[EMB + c], s2);
}

__global__ void bn_finalize(const float* __restrict__ accum, const float* __restrict__ g,
                            const float* __restrict__ b, float* __restrict__ ss,
                            int c_dim, float invn) {
    int c = blockIdx.x * blockDim.x + threadIdx.x;
    if (c >= c_dim) return;
    float m   = accum[c] * invn;
    float var = accum[c_dim + c] * invn - m * m;
    float sc  = g[c] * rsqrtf(var + BN_EPS);
    ss[c]         = sc;
    ss[c_dim + c] = b[c] - m * sc;
}

// ---------------------------------------------------------------------------
// fused BN + ReLU + mean/max pool.  gmax via int-bit atomicMax (valid: v >= 0)
__global__ void pool_kernel(const float* __restrict__ agg, const float* __restrict__ ss,
                            const int* __restrict__ batch, float* __restrict__ gsum,
                            float* __restrict__ gmax, float* __restrict__ cnt, int n) {
    long long t = (long long)blockIdx.x * blockDim.x + threadIdx.x;
    int r = (int)(t >> 6);
    if (r >= n) return;
    int c = (int)(t & 63);
    float v = fmaxf(agg[(size_t)r * EMB + c] * ss[c] + ss[EMB + c], 0.f);
    int b = batch[r];
    atomicAdd(&gsum[(size_t)b * EMB + c], v);
    atomicMax((int*)&gmax[(size_t)b * EMB + c], __float_as_int(v));
    if (c == 0) atomicAdd(&cnt[b], 1.0f);
}

__global__ void pool_finalize(const float* __restrict__ gsum, const float* __restrict__ gmax,
                              const float* __restrict__ cnt, float* __restrict__ z) {
    int t = blockIdx.x * blockDim.x + threadIdx.x;    // G * 128
    if (t >= GG * 2 * EMB) return;
    int g = t >> 7, j = t & 127;
    if (j < EMB) z[t] = gsum[g * EMB + j] / fmaxf(cnt[g], 1.0f);
    else         z[t] = gmax[g * EMB + (j - EMB)];    // init 0 handles empty graphs
}

// ---------------------------------------------------------------------------
// one MLP layer on [256, fin] -> [256, fout], optional training-mode BN + ReLU.
// Single workgroup (1024 threads) so batch stats can be computed with barriers.
// Pre-BN activations live in `out` and are rewritten elementwise (safe).
__global__ __launch_bounds__(1024) void head_layer(const float* __restrict__ in,
                                                   const float* __restrict__ w,
                                                   const float* __restrict__ bias,
                                                   const float* __restrict__ gamma,
                                                   const float* __restrict__ beta,
                                                   float* __restrict__ out,
                                                   int fin, int fout, int bn_relu) {
    __shared__ float sscale[256], sshift[256];
    int tid = threadIdx.x;
    int total = GG * fout;
    for (int o = tid; o < total; o += 1024) {
        int g = o / fout, j = o - g * fout;
        const float* ir = in + (size_t)g * fin;
        float s = bias[j];
        for (int k = 0; k < fin; ++k) s += ir[k] * w[k * fout + j];
        out[o] = s;
    }
    if (!bn_relu) return;
    __syncthreads();
    if (tid < fout) {
        float s = 0.f, s2 = 0.f;
        for (int g = 0; g < GG; ++g) { float v = out[g * fout + tid]; s += v; s2 += v * v; }
        float m   = s * (1.0f / GG);
        float var = s2 * (1.0f / GG) - m * m;
        float sc  = gamma[tid] * rsqrtf(var + BN_EPS);
        sscale[tid] = sc;
        sshift[tid] = beta[tid] - m * sc;
    }
    __syncthreads();
    for (int o = tid; o < total; o += 1024) {
        int j = o % fout;
        out[o] = fmaxf(out[o] * sscale[j] + sshift[j], 0.f);
    }
}

// ---------------------------------------------------------------------------
static inline int cdiv(long long a, int b) { return (int)((a + b - 1) / b); }

extern "C" void kernel_launch(void* const* d_in, const int* in_sizes, int n_in,
                              void* d_out, int out_size, void* d_ws, size_t ws_size,
                              hipStream_t stream) {
    const float* x      = (const float*)d_in[0];
    const int*   ei     = (const int*)d_in[1];
    const int*   batch  = (const int*)d_in[2];
    const float* conv_w = (const float*)d_in[3];
    const float* conv_b = (const float*)d_in[4];
    const float* bnc_g  = (const float*)d_in[5];
    const float* bnc_b  = (const float*)d_in[6];
    const float* fc1_w  = (const float*)d_in[7];
    const float* fc1_b  = (const float*)d_in[8];
    const float* bn1_g  = (const float*)d_in[9];
    const float* bn1_b  = (const float*)d_in[10];
    const float* fc2_w  = (const float*)d_in[11];
    const float* fc2_b  = (const float*)d_in[12];
    const float* bn2_g  = (const float*)d_in[13];
    const float* bn2_b  = (const float*)d_in[14];
    const float* fc3_w  = (const float*)d_in[15];
    const float* fc3_b  = (const float*)d_in[16];
    const float* bn3_g  = (const float*)d_in[17];
    const float* bn3_b  = (const float*)d_in[18];
    const float* out_w  = (const float*)d_in[19];
    const float* out_b  = (const float*)d_in[20];
    float* out = (float*)d_out;

    const int N = in_sizes[0] / FIN;
    const int E = in_sizes[1] / 2;

    // workspace layout (floats)
    float* ws   = (float*)d_ws;
    float* h    = ws;                         // N*64
    float* agg  = h    + (size_t)N * EMB;     // N*64
    float* dinv = agg  + (size_t)N * EMB;     // N
    float* acc  = dinv + N;                   // 128  (bn accum)
    float* ss   = acc  + 2 * EMB;             // 128  (bn scale/shift)
    float* gsum = ss   + 2 * EMB;             // G*64
    float* gmax = gsum + (size_t)GG * EMB;    // G*64
    float* cnt  = gmax + (size_t)GG * EMB;    // G
    float* z    = cnt  + GG;                  // G*128
    float* z1   = z    + (size_t)GG * 2 * EMB;// G*256
    float* z2   = z1   + (size_t)GG * 256;    // G*128
    float* z3   = z2   + (size_t)GG * 128;    // G*64

    // ---- degree / dinv -----------------------------------------------------
    fill_f32<<<cdiv(N, 256), 256, 0, stream>>>(dinv, N, 0.0f);
    count_deg<<<cdiv(E, 256), 256, 0, stream>>>(ei, dinv, E);
    finalize_dinv<<<cdiv(N, 256), 256, 0, stream>>>(dinv, N);

    // ---- h = x @ conv_w (WMMA) --------------------------------------------
    conv_gemm_wmma<<<cdiv(N, 128), 256, 0, stream>>>(x, conv_w, h, N);

    // ---- aggregation -------------------------------------------------------
    init_agg<<<cdiv((long long)N * EMB, 256), 256, 0, stream>>>(h, dinv, conv_b, agg, N);
    edge_scatter<<<cdiv((long long)E * 16, 256), 256, 0, stream>>>(ei, h, dinv, agg, E);

    // ---- BN stats + fused BN/ReLU/pool ------------------------------------
    fill_f32<<<1, 256, 0, stream>>>(acc, 2 * EMB, 0.0f);
    bn_stats<<<256, 256, 0, stream>>>(agg, acc, N);
    bn_finalize<<<1, 64, 0, stream>>>(acc, bnc_g, bnc_b, ss, EMB, 1.0f / (float)N);

    long long poolbuf = (long long)GG * EMB * 2 + GG;   // gsum|gmax|cnt contiguous
    fill_f32<<<cdiv(poolbuf, 256), 256, 0, stream>>>(gsum, poolbuf, 0.0f);
    pool_kernel<<<cdiv((long long)N * EMB, 256), 256, 0, stream>>>(agg, ss, batch, gsum, gmax, cnt, N);
    pool_finalize<<<cdiv(GG * 2 * EMB, 256), 256, 0, stream>>>(gsum, gmax, cnt, z);

    // ---- MLP head (single-block layers, training-mode BN) ------------------
    head_layer<<<1, 1024, 0, stream>>>(z,  fc1_w, fc1_b, bn1_g, bn1_b, z1, 2 * EMB, 256, 1);
    head_layer<<<1, 1024, 0, stream>>>(z1, fc2_w, fc2_b, bn2_g, bn2_b, z2, 256, 128, 1);
    head_layer<<<1, 1024, 0, stream>>>(z2, fc3_w, fc3_b, bn3_g, bn3_b, z3, 128, 64, 1);
    head_layer<<<1, 1024, 0, stream>>>(z3, out_w, out_b, nullptr, nullptr, out, 64, NC, 0);
}
